// Encoder_56410100466232
// MI455X (gfx1250) — matrix operand order repaired and checked by reading, not compiled, not run on hardware
//
#include <hip/hip_runtime.h>
#include <hip/hip_bf16.h>

// ---------------------------------------------------------------------------
// Problem constants (from the reference)
// ---------------------------------------------------------------------------
#define BB 16
#define SS 512
#define HH 768
#define AH 12
#define FF 3072
#define DH 64
#define NROW (BB * SS)   // 8192

typedef __attribute__((ext_vector_type(16))) __bf16 v16bf;
typedef __attribute__((ext_vector_type(8)))  float  v8f;

union Frag {
    v16bf v;
    unsigned short u[16];
};

__device__ __forceinline__ unsigned short f2bf(float f) {
    unsigned int u = __float_as_uint(f);
    u += 0x7FFFu + ((u >> 16) & 1u);      // round-to-nearest-even
    return (unsigned short)(u >> 16);
}

__device__ __forceinline__ v8f v8f_zero() {
    v8f z;
#pragma unroll
    for (int i = 0; i < 8; ++i) z[i] = 0.0f;
    return z;
}

// gfx1250 async global->LDS copy (16 bytes per lane), tracked by ASYNCcnt.
__device__ __forceinline__ void async_copy16(const void* g, void* lds) {
    unsigned lds_off = (unsigned)(size_t)lds;              // LDS addr = addr[31:0]
    unsigned long long ga = (unsigned long long)(size_t)g;
    asm volatile("global_load_async_to_lds_b128 %0, %1, off"
                 :: "v"(lds_off), "v"(ga)
                 : "memory");
}
__device__ __forceinline__ void wait_async0() {
    asm volatile("s_wait_asynccnt 0" ::: "memory");
}

// A-matrix fragment (16x32 bf16): lane<16 -> row=lane, K={0..7,16..23};
// lane>=16 -> row=lane-16, K pattern shifted by +8. (ISA 7.12.2)
__device__ __forceinline__ void load_a_frag(const unsigned short* base, int ld,
                                            int kbase, int lane, Frag& f) {
    const unsigned short* p = base + (lane & 15) * ld + kbase + (((lane >> 4) & 1) << 3);
#pragma unroll
    for (int j = 0; j < 4; ++j) {
        f.u[2 * j]     = p[2 * j];
        f.u[2 * j + 1] = p[2 * j + 1];
    }
#pragma unroll
    for (int j = 4; j < 8; ++j) {
        int k = 16 + 2 * (j - 4);
        f.u[2 * j]     = p[k];
        f.u[2 * j + 1] = p[k + 1];
    }
}

// B-matrix fragment (32x16 bf16): lane<16 -> col=lane, K={0..7,16..23};
// lane>=16 -> col=lane-16, K pattern shifted by +8. LDS holds B transposed: [n][k].
__device__ __forceinline__ void load_b_frag(const unsigned short* base, int ld,
                                            int nbase, int kbase, int lane, Frag& f) {
    const unsigned short* p = base + (nbase + (lane & 15)) * ld + kbase + (((lane >> 4) & 1) << 3);
#pragma unroll
    for (int j = 0; j < 16; ++j) f.u[j] = p[j < 8 ? j : j + 8];
}

__device__ __forceinline__ v8f wmma_bf16(const Frag& a, const Frag& b, v8f c) {
    return __builtin_amdgcn_wmma_f32_16x16x32_bf16(false, a.v, false, b.v,
                                                   (short)0, c, false, false);
}

__device__ __forceinline__ float gelu_exact(float x) {
    return 0.5f * x * (1.0f + erff(x * 0.70710678118654752440f));
}

__device__ __forceinline__ float halfmax16(float v) {
#pragma unroll
    for (int m = 1; m < 16; m <<= 1) v = fmaxf(v, __shfl_xor(v, m, 32));
    return v;
}
__device__ __forceinline__ float halfsum16(float v) {
#pragma unroll
    for (int m = 1; m < 16; m <<= 1) v += __shfl_xor(v, m, 32);
    return v;
}

// ---------------------------------------------------------------------------
// Generic GEMM: C[M,N] = act(A[M,K] @ W[K,N] + bias[N])   (compile-time modes)
//   ABF16: 0 -> A fp32 (convert while staging); 1 -> A bf16 (async LDS copy)
//   OUTM : 0 -> fp32 row-major; 1 -> bf16 row-major;
//          2 -> bf16 "V-transposed"  C16[(b*H + feature)*S + s]  (N must be 768)
//   ACT  : 1 -> exact GELU
// 128x128 block tile, BK=32, 8 waves (2x4), wave tile 64x32 (eight 16x16 acc).
// ---------------------------------------------------------------------------
template <int ABF16, int OUTM, int ACT>
__global__ __launch_bounds__(256) void gemm_kernel(
    const void* __restrict__ Av, const float* __restrict__ W,
    const float* __restrict__ bias, void* __restrict__ Cv,
    int N, int Kdim) {
    constexpr int LDT = 48;                       // ushorts; 96B row -> 16B aligned
    __shared__ unsigned short sA[128 * LDT];
    __shared__ unsigned short sB[128 * LDT];

    const int tid  = threadIdx.x;
    const int lane = tid & 31;
    const int wave = tid >> 5;
    const int wm = wave >> 2;                     // 0..1
    const int wn = wave & 3;                      // 0..3
    const int rowBase = blockIdx.y * 128;
    const int colBase = blockIdx.x * 128;

    // hoisted per-thread staging pointers (advance by constant stride per k-step)
    const unsigned short* aSrc16[2];
    void*                 aDst16[2];
    const float*          aSrc32[4];
    unsigned short*       aDst32[4];
    if constexpr (ABF16) {
        const unsigned short* A16 = (const unsigned short*)Av;
#pragma unroll
        for (int it = 0; it < 2; ++it) {
            int slot = tid + it * 256;            // 0..511
            int r  = slot >> 2;                   // 0..127
            int c8 = (slot & 3) << 3;             // 0,8,16,24
            aSrc16[it] = A16 + (size_t)(rowBase + r) * Kdim + c8;
            aDst16[it] = &sA[r * LDT + c8];
        }
    } else {
        const float* A32 = (const float*)Av;
#pragma unroll
        for (int it = 0; it < 4; ++it) {
            int slot = tid + it * 256;
            int r  = slot >> 3;                   // 0..127
            int cq = (slot & 7) << 2;             // 0..28
            aSrc32[it] = A32 + (size_t)(rowBase + r) * Kdim + cq;
            aDst32[it] = &sA[r * LDT + cq];
        }
    }
    const float* bSrc[4];
    unsigned short* bDst[4];
#pragma unroll
    for (int it = 0; it < 4; ++it) {
        int slot = tid + it * 256;
        int r  = slot >> 5;                       // k row 0..31
        int cq = (slot & 31) << 2;                // col 0..124
        bSrc[it] = W + (size_t)r * N + colBase + cq;
        bDst[it] = &sB[cq * LDT + r];             // +j*LDT for j-th element
    }
    const size_t bStride = (size_t)32 * N;

    v8f acc[4][2];
#pragma unroll
    for (int mi = 0; mi < 4; ++mi)
#pragma unroll
        for (int ni = 0; ni < 2; ++ni) acc[mi][ni] = v8f_zero();

    for (int kt = 0; kt < Kdim; kt += 32) {
        // ---- stage A tile 128x32 ----
        if constexpr (ABF16) {
#pragma unroll
            for (int it = 0; it < 2; ++it) async_copy16(aSrc16[it], aDst16[it]);
        } else {
#pragma unroll
            for (int it = 0; it < 4; ++it) {
                float4 v = *reinterpret_cast<const float4*>(aSrc32[it]);
                unsigned short* d = aDst32[it];
                d[0] = f2bf(v.x); d[1] = f2bf(v.y);
                d[2] = f2bf(v.z); d[3] = f2bf(v.w);
            }
        }
        // ---- stage B tile 32x128, transposed to [n][k] (fp32 weights) ----
#pragma unroll
        for (int it = 0; it < 4; ++it) {
            float4 v = *reinterpret_cast<const float4*>(bSrc[it]);
            unsigned short* d = bDst[it];
            d[0 * LDT] = f2bf(v.x);
            d[1 * LDT] = f2bf(v.y);
            d[2 * LDT] = f2bf(v.z);
            d[3 * LDT] = f2bf(v.w);
        }
        // prefetch exactly next iteration's loads
        if (kt + 32 < Kdim) {
            if constexpr (ABF16) {
#pragma unroll
                for (int it = 0; it < 2; ++it) __builtin_prefetch(aSrc16[it] + 32, 0, 0);
            } else {
#pragma unroll
                for (int it = 0; it < 4; ++it) __builtin_prefetch(aSrc32[it] + 32, 0, 0);
            }
#pragma unroll
            for (int it = 0; it < 4; ++it) __builtin_prefetch(bSrc[it] + bStride, 0, 0);
        }
        if constexpr (ABF16) wait_async0();
        __syncthreads();

        Frag a[4], bfr[2];
#pragma unroll
        for (int mi = 0; mi < 4; ++mi)
            load_a_frag(&sA[(wm * 64 + mi * 16) * LDT], LDT, 0, lane, a[mi]);
#pragma unroll
        for (int ni = 0; ni < 2; ++ni)
            load_b_frag(&sB[0], LDT, wn * 32 + ni * 16, 0, lane, bfr[ni]);
#pragma unroll
        for (int mi = 0; mi < 4; ++mi)
#pragma unroll
            for (int ni = 0; ni < 2; ++ni)
                acc[mi][ni] = wmma_bf16(a[mi], bfr[ni], acc[mi][ni]);
        __syncthreads();

        // advance staging pointers
        if constexpr (ABF16) {
#pragma unroll
            for (int it = 0; it < 2; ++it) aSrc16[it] += 32;
        } else {
#pragma unroll
            for (int it = 0; it < 4; ++it) aSrc32[it] += 32;
        }
#pragma unroll
        for (int it = 0; it < 4; ++it) bSrc[it] += bStride;
    }

    // epilogue: bias (+GELU). C layout: row r / r+8 per lane half (ISA 7.12.2).
    const int colin = lane & 15;
    const int rowo  = ((lane >> 4) & 1) << 3;
#pragma unroll
    for (int mi = 0; mi < 4; ++mi) {
        int gr = rowBase + wm * 64 + mi * 16 + rowo;
#pragma unroll
        for (int ni = 0; ni < 2; ++ni) {
            int gc = colBase + wn * 32 + ni * 16 + colin;
            float bv = bias[gc];
#pragma unroll
            for (int r = 0; r < 8; ++r) {
                float v = acc[mi][ni][r] + bv;
                if constexpr (ACT) v = gelu_exact(v);
                if constexpr (OUTM == 0) {
                    ((float*)Cv)[(size_t)(gr + r) * N + gc] = v;
                } else if constexpr (OUTM == 1) {
                    ((unsigned short*)Cv)[(size_t)(gr + r) * N + gc] = f2bf(v);
                } else {
                    int tok = gr + r;
                    int bidx = tok >> 9;          // / S
                    int s = tok & (SS - 1);
                    ((unsigned short*)Cv)[((size_t)bidx * HH + gc) * SS + s] = f2bf(v);
                }
            }
        }
    }
}

// ---------------------------------------------------------------------------
// Flash attention: one workgroup = (batch b, head h, 128-query tile).
// Q16,K16: bf16 (B,S,H). Vt16: bf16 pre-transposed [(b*H + h*64 + d)][s].
// Z16 out: bf16 (B,S,H). All staging via async global->LDS B128 copies.
// ---------------------------------------------------------------------------
__global__ __launch_bounds__(256) void flash_attn_kernel(
    const unsigned short* __restrict__ Q16, const unsigned short* __restrict__ K16,
    const unsigned short* __restrict__ Vt16, const int* __restrict__ mask,
    unsigned short* __restrict__ Z16) {
    constexpr int LDK = 80;                       // 160B row -> 16B aligned
    __shared__ unsigned short sQ[8][16][LDK];     // per-wave Q tile
    __shared__ unsigned short sP[8][16][LDK];     // per-wave softmax probs
    __shared__ unsigned short sK[64][LDK];        // K block [key][d]
    __shared__ unsigned short sV[64][LDK];        // V block [d][key]

    const int tid  = threadIdx.x;
    const int lane = tid & 31;
    const int wave = tid >> 5;
    const int qt = blockIdx.x;                    // 0..3
    const int h  = blockIdx.y;                    // 0..11
    const int b  = blockIdx.z;                    // 0..15

    const int colin = lane & 15;
    const int rowo  = ((lane >> 4) & 1) << 3;
    const int q0 = qt * 128 + wave * 16;
    const float NEG_INF = -__builtin_inff();

    // stage this wave's 16x64 Q tile (async)
#pragma unroll
    for (int it = 0; it < 4; ++it) {
        int slot = lane + it * 32;                // 0..127
        int r  = slot >> 3;                       // 0..15
        int c8 = (slot & 7) << 3;                 // 0..56
        async_copy16(Q16 + ((size_t)b * SS + q0 + r) * HH + h * DH + c8,
                     &sQ[wave][r][c8]);
    }

    // hoisted K/V staging pointers; advance per key block
    const unsigned short* kSrc[2];
    const unsigned short* vSrc[2];
    void* kDst[2];
    void* vDst[2];
#pragma unroll
    for (int it = 0; it < 2; ++it) {
        int slot = tid + it * 256;                // 0..511
        int r  = slot >> 3;                       // 0..63
        int c8 = (slot & 7) << 3;                 // 0..56
        kSrc[it] = K16 + ((size_t)b * SS + r) * HH + h * DH + c8;
        vSrc[it] = Vt16 + ((size_t)(b * HH + h * DH + r)) * SS + c8;
        kDst[it] = &sK[r][c8];
        vDst[it] = &sV[r][c8];
    }
    const size_t kStride = (size_t)64 * HH;       // 64 key rows
    const size_t vStride = 64;                    // 64 key columns

    float m_r[8], l_r[8];
    v8f o[4];
#pragma unroll
    for (int r = 0; r < 8; ++r) { m_r[r] = NEG_INF; l_r[r] = 0.0f; }
#pragma unroll
    for (int dt = 0; dt < 4; ++dt) o[dt] = v8f_zero();

    const int* maskrow = mask + (size_t)b * SS;

    for (int kb = 0; kb < SS / 64; ++kb) {
        // stage K block [key][d] and V block [d][key] (plain async copies)
#pragma unroll
        for (int it = 0; it < 2; ++it) {
            async_copy16(kSrc[it], kDst[it]);
            async_copy16(vSrc[it], vDst[it]);
        }
        // prefetch next key block (exact next-iteration addresses)
        if (kb + 1 < SS / 64) {
#pragma unroll
            for (int it = 0; it < 2; ++it) {
                __builtin_prefetch(kSrc[it] + kStride, 0, 0);
                __builtin_prefetch(vSrc[it] + vStride, 0, 0);
            }
        }
        wait_async0();
        __syncthreads();

        // S = Q @ K^T  (16 x 64 per wave)
        Frag aq[2];
        load_a_frag(&sQ[wave][0][0], LDK, 0, lane, aq[0]);
        load_a_frag(&sQ[wave][0][0], LDK, 32, lane, aq[1]);
        v8f sc[4];
#pragma unroll
        for (int nt = 0; nt < 4; ++nt) {
            v8f s = v8f_zero();
#pragma unroll
            for (int kk = 0; kk < 2; ++kk) {
                Frag bk;
                load_b_frag(&sK[0][0], LDK, nt * 16, kk * 32, lane, bk);
                s = wmma_bf16(aq[kk], bk, s);
            }
            sc[nt] = s;
        }

        // scale + mask + block row-max
        float bm[8];
#pragma unroll
        for (int r = 0; r < 8; ++r) bm[r] = NEG_INF;
#pragma unroll
        for (int nt = 0; nt < 4; ++nt) {
            int key = kb * 64 + nt * 16 + colin;
            float madd = (maskrow[key] == 0) ? NEG_INF : 0.0f;
#pragma unroll
            for (int r = 0; r < 8; ++r) {
                float vv = sc[nt][r] * 0.125f + madd;   // 1/sqrt(64)
                sc[nt][r] = vv;
                bm[r] = fmaxf(bm[r], vv);
            }
        }

        // online softmax update (rows 0-7 in lanes 0-15, rows 8-15 in 16-31)
        float corr[8], sub[8];
#pragma unroll
        for (int r = 0; r < 8; ++r) {
            float rm = halfmax16(bm[r]);
            float nm = fmaxf(m_r[r], rm);
            sub[r]  = (nm == NEG_INF) ? 0.0f : nm;
            corr[r] = __expf(m_r[r] - sub[r]);          // 0 if m was -inf (l=o=0)
            m_r[r]  = nm;
        }
        float lsum[8];
#pragma unroll
        for (int r = 0; r < 8; ++r) lsum[r] = 0.0f;
#pragma unroll
        for (int nt = 0; nt < 4; ++nt) {
#pragma unroll
            for (int r = 0; r < 8; ++r) {
                float p = __expf(sc[nt][r] - sub[r]);
                lsum[r] += p;
                sP[wave][rowo + r][nt * 16 + colin] = f2bf(p);
            }
        }
#pragma unroll
        for (int r = 0; r < 8; ++r) {
            l_r[r] = l_r[r] * corr[r] + halfsum16(lsum[r]);
#pragma unroll
            for (int dt = 0; dt < 4; ++dt) o[dt][r] *= corr[r];
        }

        // O += P @ V   (16 x 64 per wave)
#pragma unroll
        for (int kk = 0; kk < 2; ++kk) {
            Frag ap;
            load_a_frag(&sP[wave][0][0], LDK, kk * 32, lane, ap);
#pragma unroll
            for (int dt = 0; dt < 4; ++dt) {
                Frag bv;
                load_b_frag(&sV[0][0], LDK, dt * 16, kk * 32, lane, bv);
                o[dt] = wmma_bf16(ap, bv, o[dt]);
            }
        }
        __syncthreads();   // before next K/V restage

#pragma unroll
        for (int it = 0; it < 2; ++it) {
            kSrc[it] += kStride;
            vSrc[it] += vStride;
        }
    }

    // finalize: divide by row sum, write Z (B,S,H) as bf16
#pragma unroll
    for (int r = 0; r < 8; ++r) {
        float inv = 1.0f / l_r[r];
        size_t rowoff = ((size_t)b * SS + q0 + rowo + r) * HH + h * DH;
#pragma unroll
        for (int dt = 0; dt < 4; ++dt)
            Z16[rowoff + dt * 16 + colin] = f2bf(o[dt][r] * inv);
    }
}

// ---------------------------------------------------------------------------
// out = LayerNorm(x + res) * w + b   (one block per row of 768, fp32)
// ---------------------------------------------------------------------------
__global__ __launch_bounds__(256) void residual_ln_kernel(
    const float* __restrict__ x, const float* __restrict__ res,
    const float* __restrict__ w, const float* __restrict__ bias,
    float* __restrict__ out) {
    __shared__ float red[256];
    const int row = blockIdx.x;
    const int t = threadIdx.x;
    const float* xr = x + (size_t)row * HH;
    const float* rr = res + (size_t)row * HH;

    float v0 = xr[t] + rr[t];
    float v1 = xr[t + 256] + rr[t + 256];
    float v2 = xr[t + 512] + rr[t + 512];

    red[t] = v0 + v1 + v2;
    __syncthreads();
    for (int o = 128; o > 0; o >>= 1) {
        if (t < o) red[t] += red[t + o];
        __syncthreads();
    }
    float mu = red[0] * (1.0f / 768.0f);
    __syncthreads();

    float d0 = v0 - mu, d1 = v1 - mu, d2 = v2 - mu;
    red[t] = d0 * d0 + d1 * d1 + d2 * d2;
    __syncthreads();
    for (int o = 128; o > 0; o >>= 1) {
        if (t < o) red[t] += red[t + o];
        __syncthreads();
    }
    float rstd = rsqrtf(red[0] * (1.0f / 768.0f) + 1e-5f);

    float* orow = out + (size_t)row * HH;
    orow[t]       = d0 * rstd * w[t]       + bias[t];
    orow[t + 256] = d1 * rstd * w[t + 256] + bias[t + 256];
    orow[t + 512] = d2 * rstd * w[t + 512] + bias[t + 512];
}

// ---------------------------------------------------------------------------
// Orchestration
// ---------------------------------------------------------------------------
extern "C" void kernel_launch(void* const* d_in, const int* in_sizes, int n_in,
                              void* d_out, int out_size, void* d_ws, size_t ws_size,
                              hipStream_t stream) {
    (void)in_sizes; (void)n_in; (void)out_size; (void)ws_size;

    const float* X    = (const float*)d_in[0];
    const int*   mask = (const int*)d_in[1];
    const float* WQ = (const float*)d_in[2];
    const float* bQ = (const float*)d_in[3];
    const float* WK = (const float*)d_in[4];
    const float* bK = (const float*)d_in[5];
    const float* WV = (const float*)d_in[6];
    const float* bV = (const float*)d_in[7];
    const float* WO = (const float*)d_in[8];
    const float* bO = (const float*)d_in[9];
    const float* ln1w = (const float*)d_in[10];
    const float* ln1b = (const float*)d_in[11];
    const float* W1 = (const float*)d_in[12];
    const float* b1 = (const float*)d_in[13];
    const float* W2 = (const float*)d_in[14];
    const float* b2 = (const float*)d_in[15];
    const float* ln2w = (const float*)d_in[16];
    const float* ln2b = (const float*)d_in[17];
    float* out = (float*)d_out;

    const size_t nh = (size_t)NROW * HH;           // 6,291,456
    unsigned short* Q16  = (unsigned short*)d_ws;
    unsigned short* K16  = Q16 + nh;
    unsigned short* Vt16 = K16 + nh;
    unsigned short* Z16  = Vt16 + nh;
    unsigned short* H116 = Z16 + nh;               // NROW x FF (bf16)
    float* AO = (float*)(H116 + (size_t)NROW * FF);
    float* X1 = AO + nh;
    float* F2 = X1 + nh;

    dim3 blk(256);
    dim3 gH(HH / 128, NROW / 128);                 // (6, 64)
    dim3 gF(FF / 128, NROW / 128);                 // (24, 64)

    // QKV projections: fp32 in, bf16 out (V pre-transposed for attention)
    gemm_kernel<0, 1, 0><<<gH, blk, 0, stream>>>(X, WQ, bQ, Q16,  HH, HH);
    gemm_kernel<0, 1, 0><<<gH, blk, 0, stream>>>(X, WK, bK, K16,  HH, HH);
    gemm_kernel<0, 2, 0><<<gH, blk, 0, stream>>>(X, WV, bV, Vt16, HH, HH);

    flash_attn_kernel<<<dim3(SS / 128, AH, BB), blk, 0, stream>>>(Q16, K16, Vt16, mask, Z16);

    // O-projection: bf16 in, fp32 out
    gemm_kernel<1, 0, 0><<<gH, blk, 0, stream>>>(Z16, WO, bO, AO, HH, HH);
    residual_ln_kernel<<<NROW, blk, 0, stream>>>(AO, X, ln1w, ln1b, X1);

    // FFN: fp32 in -> GELU -> bf16 H1 -> bf16 in -> fp32 out
    gemm_kernel<0, 1, 1><<<gF, blk, 0, stream>>>(X1, W1, b1, H116, FF, HH);
    gemm_kernel<1, 0, 0><<<gH, blk, 0, stream>>>(H116, W2, b2, F2, HH, FF);
    residual_ln_kernel<<<NROW, blk, 0, stream>>>(F2, X1, ln2w, ln2b, out);
}
